// SGPN_45526653337883
// MI455X (gfx1250) — matrix operand order repaired and checked by reading, not compiled, not run on hardware
//
#include <hip/hip_runtime.h>
#include <hip/hip_bf16.h>

typedef __attribute__((ext_vector_type(8)))  float    v8f;
typedef __attribute__((ext_vector_type(8)))  _Float16 v8h;
typedef __attribute__((ext_vector_type(16))) _Float16 v16h;

#define N_NODES 512
#define N_EDGES 4096
#define DD      256
#define DD3     768
#define HG      512
#define NOC     160
#define NRC     26

#define USE_ASYNC 1

// ---------------- WMMA helpers (CDNA5 wave32, 16x16x32 f16 -> f32) -------------

__device__ __forceinline__ v8f wmma_f16(v16h a, v16h b, v8f c) {
  return __builtin_amdgcn_wmma_f32_16x16x32_f16(false, a, false, b, (short)0, c,
                                                false, false);
}

__device__ __forceinline__ void dsfence() {
  asm volatile("s_wait_dscnt 0" ::: "memory");
}

#if USE_ASYNC
// GLOBAL_LOAD_ASYNC_TO_LDS_B128: vdst = 32-bit LDS byte address (generic LDS
// pointers map to LDS offset via addr[31:0] truncation per the aperture rule),
// vaddr = 64-bit global address, saddr = off (GV mode). Tracked by ASYNCcnt.
__device__ __forceinline__ void async_copy16(void* lds, const void* g) {
  asm volatile("global_load_async_to_lds_b128 %0, %1, off"
               :
               : "v"((uint32_t)(size_t)lds), "v"((unsigned long long)(size_t)g)
               : "memory");
}
__device__ __forceinline__ void wait_async() {
  asm volatile("s_wait_asynccnt 0" ::: "memory");
}
#else
__device__ __forceinline__ void async_copy16(void* lds, const void* g) {
  *(uint4*)lds = *(const uint4*)g;
}
__device__ __forceinline__ void wait_async() {}
#endif

__device__ __forceinline__ v16h cat8(v8h a, v8h b) {
  return __builtin_shufflevector(a, b, 0, 1, 2, 3, 4, 5, 6, 7, 8, 9, 10, 11, 12,
                                 13, 14, 15);
}

// A fragment: 16x32 f16 tile, row-major, leading dim lda (elements, mult of 16).
// Lane L: m = L&15, kb = (L>>4)*8; halves = [kb..kb+7] ++ [16+kb..16+kb+7]
// -> two contiguous 16B runs -> 2x ds_load_b128.
__device__ __forceinline__ v16h load_a_frag(const _Float16* Ab, int lda, int koff,
                                            int lane) {
  const _Float16* p = &Ab[(lane & 15) * lda + koff + ((lane >> 4) << 3)];
  return cat8(*(const v8h*)p, *(const v8h*)(p + 16));
}

// B fragment from TRANSPOSED weights BT[N][K] (ldk mult of 16).
// Lane L: n = noff + (L&15), kb = (L>>4)*8; halves = [kb..kb+7] ++ [16+kb..16+kb+7].
__device__ __forceinline__ v16h load_bT_frag(const _Float16* BT, int ldk, int koff,
                                             int noff, int lane) {
  const _Float16* p = &BT[(size_t)(noff + (lane & 15)) * ldk + koff +
                          ((lane >> 4) << 3)];
  return cat8(*(const v8h*)p, *(const v8h*)(p + 16));
}

// ---------------- Fused PointNet: [B][CIN][NPTS] -> max over pts of MLP --------
// Block = 256 thr (8 waves); one block per element; wave owns 16-point tiles.
// Weights arrive pre-transposed f16: w1T[64][32], w2T[128][64], w3T[256][128].
template <int CIN, int NPTS>
__global__ void pointnet_kernel(const float* __restrict__ pts,
                                const _Float16* __restrict__ w1T, const float* __restrict__ b1,
                                const _Float16* __restrict__ w2T, const float* __restrict__ b2,
                                const _Float16* __restrict__ w3T, const float* __restrict__ b3,
                                _Float16* __restrict__ outH) {
  extern __shared__ char smem[];
  _Float16* sW2 = (_Float16*)smem;          // [128][64]  (T)
  _Float16* sW3 = sW2 + 128 * 64;           // [256][128] (T)
  _Float16* sW1 = sW3 + 256 * 128;          // [64][32]   (T, zero-padded K)
  _Float16* sStage = sW1 + 64 * 32;         // 8 waves x [16][128]
  _Float16* sXt = sStage + 8 * 16 * 128;    // 8 waves x [16][32]
  float* sB1 = (float*)(sXt + 8 * 16 * 32); // 64
  float* sB2 = sB1 + 64;                    // 128
  float* sB3 = sB2 + 128;                   // 256
  float* sMax = sB3 + 256;                  // 8 x 256

  const int tid = threadIdx.x;
  const int lane = tid & 31;
  const int w = tid >> 5;
  const int b = blockIdx.x;
  const int m16 = lane & 15;
  const int hi = lane >> 4;

  // weight staging: async 16B global->LDS copies
  for (int i = tid; i < 128 * 64 / 8; i += 256) async_copy16(&sW2[i * 8], &w2T[i * 8]);
  for (int i = tid; i < 256 * 128 / 8; i += 256) async_copy16(&sW3[i * 8], &w3T[i * 8]);
  for (int i = tid; i < 64 * 32 / 8; i += 256) async_copy16(&sW1[i * 8], &w1T[i * 8]);
  for (int i = tid; i < 64; i += 256) sB1[i] = b1[i];
  for (int i = tid; i < 128; i += 256) sB2[i] = b2[i];
  for (int i = tid; i < 256; i += 256) sB3[i] = b3[i];
  wait_async();
  __syncthreads();

  _Float16* stg = sStage + w * 16 * 128;
  _Float16* xt  = sXt + w * 16 * 32;

  float tmax[16];
#pragma unroll
  for (int i = 0; i < 16; ++i) tmax[i] = -3.0e38f;

  for (int t = w; t < NPTS / 16; t += 8) {
    // stage X tile [16][32] f16, zero-padded K
#pragma unroll
    for (int j = 0; j < 16; ++j) {
      int idx = lane * 16 + j;
      int m = idx >> 5, k = idx & 31;
      float v = 0.f;
      if (k < CIN) v = pts[(size_t)b * CIN * NPTS + k * NPTS + t * 16 + m];
      xt[m * 32 + k] = (_Float16)v;
    }
    dsfence();
    // layer 1: [16x32(pad)] @ W1 -> relu -> stage [16][64]
    {
      v16h a = load_a_frag(xt, 32, 0, lane);
#pragma unroll
      for (int nc = 0; nc < 4; ++nc) {
        float bias = sB1[nc * 16 + m16];
        v8f acc;
#pragma unroll
        for (int r = 0; r < 8; ++r) acc[r] = bias;
        acc = wmma_f16(a, load_bT_frag(sW1, 32, 0, nc * 16, lane), acc);
#pragma unroll
        for (int r = 0; r < 8; ++r) {
          float v = acc[r] > 0.f ? acc[r] : 0.f;
          stg[(hi * 8 + r) * 64 + nc * 16 + m16] = (_Float16)v;
        }
      }
    }
    dsfence();
    // layer 2: [16x64] @ W2 -> relu -> stage [16][128]
    {
      v16h a0 = load_a_frag(stg, 64, 0, lane);
      v16h a1 = load_a_frag(stg, 64, 32, lane);
#pragma unroll
      for (int nc = 0; nc < 8; ++nc) {
        float bias = sB2[nc * 16 + m16];
        v8f acc;
#pragma unroll
        for (int r = 0; r < 8; ++r) acc[r] = bias;
        acc = wmma_f16(a0, load_bT_frag(sW2, 64, 0, nc * 16, lane), acc);
        acc = wmma_f16(a1, load_bT_frag(sW2, 64, 32, nc * 16, lane), acc);
#pragma unroll
        for (int r = 0; r < 8; ++r) {
          float v = acc[r] > 0.f ? acc[r] : 0.f;
          stg[(hi * 8 + r) * 128 + nc * 16 + m16] = (_Float16)v;
        }
      }
    }
    dsfence();
    // layer 3: [16x128] @ W3 + b3, column-max over the 16 points
    {
      v16h a3[4];
#pragma unroll
      for (int kc = 0; kc < 4; ++kc) a3[kc] = load_a_frag(stg, 128, kc * 32, lane);
#pragma unroll
      for (int nc = 0; nc < 16; ++nc) {
        float bias = sB3[nc * 16 + m16];
        v8f acc;
#pragma unroll
        for (int r = 0; r < 8; ++r) acc[r] = bias;
#pragma unroll
        for (int kc = 0; kc < 4; ++kc)
          acc = wmma_f16(a3[kc], load_bT_frag(sW3, 128, kc * 32, nc * 16, lane), acc);
        float cm = acc[0];
#pragma unroll
        for (int r = 1; r < 8; ++r) cm = fmaxf(cm, acc[r]);
        cm = fmaxf(cm, __shfl_xor(cm, 16, 32));  // combine m=0..7 with m=8..15
        tmax[nc] = fmaxf(tmax[nc], cm);
      }
    }
  }
  if (lane < 16) {
#pragma unroll
    for (int nc = 0; nc < 16; ++nc) sMax[w * 256 + nc * 16 + lane] = tmax[nc];
  }
  __syncthreads();
  {
    int col = tid;  // 256 cols, 256 threads
    float m = sMax[col];
#pragma unroll
    for (int ww = 1; ww < 8; ++ww) m = fmaxf(m, sMax[ww * 256 + col]);
    outH[(size_t)b * 256 + col] = (_Float16)m;
  }
}

// ---------------- Generic f16 WMMA GEMM: C = act(A[MxK] @ BT[NxK]^T + bias) ----
// B is pre-transposed [N][K]. Block tile 128x128, 8 waves (2 M-tiles x 4 N-tiles
// each), K-step 32. Tile staging via async global->LDS; 12 ds_load_b128 -> 8 wmma
// per wave per K-step.
__global__ void gemm_kernel(const _Float16* __restrict__ A, const _Float16* __restrict__ BT,
                            const float* __restrict__ bias,
                            float* __restrict__ Cf, _Float16* __restrict__ Ch,
                            int M, int N, int K, int act) {
  __shared__ __align__(16) _Float16 As[128 * 32];
  __shared__ __align__(16) _Float16 BsT[128 * 32];  // [n][k] within tile
  const int tid = threadIdx.x;
  const int lane = tid & 31;
  const int w = tid >> 5;
  const int m0 = blockIdx.x * 128;
  const int n0 = blockIdx.y * 128;
  const int mloc = (w & 3) * 32;   // two 16-row tiles: mloc, mloc+16
  const int nloc = (w >> 2) * 64;  // four 16-col tiles
  const int m16 = lane & 15;
  const int hi = lane >> 4;

  v8f acc[2][4];
#pragma unroll
  for (int j = 0; j < 4; ++j) {
    int col = n0 + nloc + j * 16 + m16;
    float bv = (col < N) ? bias[col] : 0.f;
#pragma unroll
    for (int mt = 0; mt < 2; ++mt)
#pragma unroll
      for (int r = 0; r < 8; ++r) acc[mt][j][r] = bv;
  }

  const int rr = tid >> 1;          // 0..127
  const int ks = (tid & 1) * 16;    // 0 or 16

  for (int k0 = 0; k0 < K; k0 += 32) {
    __syncthreads();
    {  // A tile 128x32 (K mult of 32, M mult of 128): async 16B copies
      const _Float16* src = &A[(size_t)(m0 + rr) * K + k0 + ks];
      async_copy16(&As[rr * 32 + ks], src);
      async_copy16(&As[rr * 32 + ks + 8], src + 8);
      if (k0 + 32 < K) __builtin_prefetch(src + 32, 0, 1);
    }
    {  // B tile: rows n0+rr of BT, zero-padded past N
      int col = n0 + rr;
      if (col < N) {
        const _Float16* src = &BT[(size_t)col * K + k0 + ks];
        async_copy16(&BsT[rr * 32 + ks], src);
        async_copy16(&BsT[rr * 32 + ks + 8], src + 8);
      } else {
        uint4 z = {0u, 0u, 0u, 0u};
        *(uint4*)&BsT[rr * 32 + ks] = z;
        *(uint4*)&BsT[rr * 32 + ks + 8] = z;
      }
    }
    wait_async();
    __syncthreads();
    v16h a0 = load_a_frag(&As[mloc * 32], 32, 0, lane);
    v16h a1 = load_a_frag(&As[(mloc + 16) * 32], 32, 0, lane);
#pragma unroll
    for (int j = 0; j < 4; ++j) {
      v16h bf = load_bT_frag(BsT, 32, 0, nloc + j * 16, lane);
      acc[0][j] = wmma_f16(a0, bf, acc[0][j]);
      acc[1][j] = wmma_f16(a1, bf, acc[1][j]);
    }
  }

#pragma unroll
  for (int mt = 0; mt < 2; ++mt) {
#pragma unroll
    for (int j = 0; j < 4; ++j) {
#pragma unroll
      for (int r = 0; r < 8; ++r) {
        int row = m0 + mloc + mt * 16 + hi * 8 + r;
        int col = n0 + nloc + j * 16 + m16;
        if (col < N) {
          float v = acc[mt][j][r];
          if (act) v = v > 0.f ? v : 0.f;
          if (Cf) Cf[(size_t)row * N + col] = v;
          if (Ch) Ch[(size_t)row * N + col] = (_Float16)v;
        }
      }
    }
  }
}

// ---------------- Aux kernels --------------------------------------------------

// f32 [K][N] -> f16 transposed [N][Kpad], zero-padded K..Kpad
__global__ void cvt_t_kernel(const float* __restrict__ src, _Float16* __restrict__ dst,
                             int K, int N, int Kpad) {
  int i = blockIdx.x * 256 + threadIdx.x;
  if (i < N * Kpad) {
    int n = i / Kpad, k = i % Kpad;
    dst[i] = (k < K) ? (_Float16)src[(size_t)k * N + n] : (_Float16)0.f;
  }
}
__global__ void zero_kernel(float* p, int n) {
  int i = blockIdx.x * 256 + threadIdx.x;
  if (i < n) p[i] = 0.f;
}
__global__ void deg_kernel(const int* __restrict__ ei, float* __restrict__ cnt) {
  int e = blockIdx.x * 256 + threadIdx.x;
  if (e < N_EDGES) {
    atomicAdd(&cnt[ei[e]], 1.f);
    atomicAdd(&cnt[ei[N_EDGES + e]], 1.f);
  }
}
__global__ void gather_kernel(const _Float16* __restrict__ xh, const _Float16* __restrict__ eh,
                              const int* __restrict__ ei, _Float16* __restrict__ tb) {
  int e = blockIdx.x;
  int s = ei[e], o = ei[N_EDGES + e];
  for (int c = threadIdx.x; c < DD3; c += 256) {
    _Float16 v;
    if (c < DD)            v = xh[(size_t)s * DD + c];
    else if (c < 2 * DD)   v = eh[(size_t)e * DD + (c - DD)];
    else                   v = xh[(size_t)o * DD + (c - 2 * DD)];
    tb[(size_t)e * DD3 + c] = v;
  }
}
__global__ void scatter_kernel(const float* __restrict__ hout, const int* __restrict__ ei,
                               _Float16* __restrict__ eh, float* __restrict__ pooled) {
  int e = blockIdx.x;
  int s = ei[e], o = ei[N_EDGES + e];
  int c = threadIdx.x;  // 256 cols
  float ns = hout[(size_t)e * DD3 + c];
  float ne = hout[(size_t)e * DD3 + DD + c];
  float no = hout[(size_t)e * DD3 + 2 * DD + c];
  atomicAdd(&pooled[(size_t)s * DD + c], ns);
  atomicAdd(&pooled[(size_t)o * DD + c], no);
  eh[(size_t)e * DD + c] = (_Float16)ne;
}
__global__ void avg_kernel(const float* __restrict__ pooled, const float* __restrict__ cnt,
                           _Float16* __restrict__ avgh) {
  int n = blockIdx.x, c = threadIdx.x;
  float d = fmaxf(cnt[n], 1.f);
  avgh[(size_t)n * DD + c] = (_Float16)(pooled[(size_t)n * DD + c] / d);
}
__global__ void logsoftmax_kernel(const float* __restrict__ lg, float* __restrict__ out) {
  __shared__ float red[256];
  int row = blockIdx.x, tid = threadIdx.x;
  float v = (tid < NOC) ? lg[(size_t)row * NOC + tid] : -3.0e38f;
  red[tid] = v;
  __syncthreads();
  for (int s = 128; s > 0; s >>= 1) {
    if (tid < s) red[tid] = fmaxf(red[tid], red[tid + s]);
    __syncthreads();
  }
  float mx = red[0];
  __syncthreads();
  red[tid] = (tid < NOC) ? __expf(v - mx) : 0.f;
  __syncthreads();
  for (int s = 128; s > 0; s >>= 1) {
    if (tid < s) red[tid] += red[tid + s];
    __syncthreads();
  }
  float lse = __logf(red[0]);
  if (tid < NOC) out[(size_t)row * NOC + tid] = v - mx - lse;
}
__global__ void sigmoid_kernel(const float* __restrict__ x, float* __restrict__ out, int n) {
  int i = blockIdx.x * 256 + threadIdx.x;
  if (i < n) out[i] = 1.f / (1.f + __expf(-x[i]));
}

// ---------------- Host orchestration ------------------------------------------

extern "C" void kernel_launch(void* const* d_in, const int* in_sizes, int n_in,
                              void* d_out, int out_size, void* d_ws, size_t ws_size,
                              hipStream_t stream) {
  const float* obj_points = (const float*)d_in[0];
  const float* rel_points = (const float*)d_in[1];
  const int*   ei         = (const int*)d_in[2];
  const float* obj_w1 = (const float*)d_in[3],  *obj_b1 = (const float*)d_in[4];
  const float* obj_w2 = (const float*)d_in[5],  *obj_b2 = (const float*)d_in[6];
  const float* obj_w3 = (const float*)d_in[7],  *obj_b3 = (const float*)d_in[8];
  const float* rel_w1 = (const float*)d_in[9],  *rel_b1 = (const float*)d_in[10];
  const float* rel_w2 = (const float*)d_in[11], *rel_b2 = (const float*)d_in[12];
  const float* rel_w3 = (const float*)d_in[13], *rel_b3 = (const float*)d_in[14];
  const float* g_tw1 = (const float*)d_in[15], *g_tb1 = (const float*)d_in[16];
  const float* g_tw2 = (const float*)d_in[17], *g_tb2 = (const float*)d_in[18];
  const float* g_nw1 = (const float*)d_in[19], *g_nb1 = (const float*)d_in[20];
  const float* g_nw2 = (const float*)d_in[21], *g_nb2 = (const float*)d_in[22];
  const float* oc_w1 = (const float*)d_in[23], *oc_b1 = (const float*)d_in[24];
  const float* oc_w2 = (const float*)d_in[25], *oc_b2 = (const float*)d_in[26];
  const float* oc_w3 = (const float*)d_in[27], *oc_b3 = (const float*)d_in[28];
  const float* rc_w1 = (const float*)d_in[29], *rc_b1 = (const float*)d_in[30];
  const float* rc_w2 = (const float*)d_in[31], *rc_b2 = (const float*)d_in[32];
  const float* rc_w3 = (const float*)d_in[33], *rc_b3 = (const float*)d_in[34];

  char* ws = (char*)d_ws;
  size_t off = 0;
  auto alloc = [&](size_t bytes) -> void* {
    void* p = (void*)(ws + off);
    off += (bytes + 255) & ~(size_t)255;
    return p;
  };

  // transposed f16 weights [N][Kpad]
  _Float16* g_tw1T = (_Float16*)alloc((size_t)2 * 512 * 768 * 2);
  _Float16* g_tw2T = (_Float16*)alloc((size_t)2 * 768 * 512 * 2);
  _Float16* g_nw1T = (_Float16*)alloc((size_t)2 * 512 * 256 * 2);
  _Float16* g_nw2T = (_Float16*)alloc((size_t)2 * 256 * 512 * 2);
  _Float16* oc_w1T = (_Float16*)alloc((size_t)512 * 256 * 2);
  _Float16* oc_w2T = (_Float16*)alloc((size_t)256 * 512 * 2);
  _Float16* oc_w3T = (_Float16*)alloc((size_t)160 * 256 * 2);
  _Float16* rc_w1T = (_Float16*)alloc((size_t)512 * 256 * 2);
  _Float16* rc_w2T = (_Float16*)alloc((size_t)256 * 512 * 2);
  _Float16* rc_w3T = (_Float16*)alloc((size_t)26 * 256 * 2);
  _Float16* ow1T = (_Float16*)alloc((size_t)64 * 32 * 2);
  _Float16* ow2T = (_Float16*)alloc((size_t)128 * 64 * 2);
  _Float16* ow3T = (_Float16*)alloc((size_t)256 * 128 * 2);
  _Float16* rw1T = (_Float16*)alloc((size_t)64 * 32 * 2);
  _Float16* rw2T = (_Float16*)alloc((size_t)128 * 64 * 2);
  _Float16* rw3T = (_Float16*)alloc((size_t)256 * 128 * 2);
  // activations
  _Float16* xh   = (_Float16*)alloc((size_t)N_NODES * DD * 2);
  _Float16* eh   = (_Float16*)alloc((size_t)N_EDGES * DD * 2);
  _Float16* tbuf = (_Float16*)alloc((size_t)N_EDGES * DD3 * 2);
  _Float16* h1h  = (_Float16*)alloc((size_t)N_EDGES * HG * 2);
  float*    hout = (float*)alloc((size_t)N_EDGES * DD3 * 4);
  float*    pooled = (float*)alloc((size_t)N_NODES * DD * 4);
  float*    cnt  = (float*)alloc((size_t)N_NODES * 4);
  _Float16* avgh = (_Float16*)alloc((size_t)N_NODES * DD * 2);
  _Float16* nh   = (_Float16*)alloc((size_t)N_NODES * HG * 2);
  _Float16* och1 = (_Float16*)alloc((size_t)N_NODES * 512 * 2);
  _Float16* och2 = (_Float16*)alloc((size_t)N_NODES * 256 * 2);
  float*    oclog = (float*)alloc((size_t)N_NODES * NOC * 4);
  _Float16* rch1 = (_Float16*)alloc((size_t)N_EDGES * 512 * 2);
  _Float16* rch2 = (_Float16*)alloc((size_t)N_EDGES * 256 * 2);
  float*    rclog = (float*)alloc((size_t)N_EDGES * NRC * 4);

  auto cvtT = [&](const float* s, _Float16* d, int K, int N, int Kpad) {
    int n = N * Kpad;
    cvt_t_kernel<<<dim3((n + 255) / 256), dim3(256), 0, stream>>>(s, d, K, N, Kpad);
  };
  auto gemm = [&](const _Float16* A, const _Float16* BT, const float* bias,
                  float* Cf, _Float16* Ch, int M, int N, int K, int act) {
    gemm_kernel<<<dim3(M / 128, (N + 127) / 128), dim3(256), 0, stream>>>(
        A, BT, bias, Cf, Ch, M, N, K, act);
  };

  // transpose + f16-convert all weights
  for (int l = 0; l < 2; ++l) {
    cvtT(g_tw1 + (size_t)l * 768 * 512, g_tw1T + (size_t)l * 512 * 768, 768, 512, 768);
    cvtT(g_tw2 + (size_t)l * 512 * 768, g_tw2T + (size_t)l * 768 * 512, 512, 768, 512);
    cvtT(g_nw1 + (size_t)l * 256 * 512, g_nw1T + (size_t)l * 512 * 256, 256, 512, 256);
    cvtT(g_nw2 + (size_t)l * 512 * 256, g_nw2T + (size_t)l * 256 * 512, 512, 256, 512);
  }
  cvtT(oc_w1, oc_w1T, 256, 512, 256);
  cvtT(oc_w2, oc_w2T, 512, 256, 512);
  cvtT(oc_w3, oc_w3T, 256, 160, 256);
  cvtT(rc_w1, rc_w1T, 256, 512, 256);
  cvtT(rc_w2, rc_w2T, 512, 256, 512);
  cvtT(rc_w3, rc_w3T, 256, 26, 256);
  cvtT(obj_w1, ow1T, 3, 64, 32);
  cvtT(obj_w2, ow2T, 64, 128, 64);
  cvtT(obj_w3, ow3T, 128, 256, 128);
  cvtT(rel_w1, rw1T, 4, 64, 32);
  cvtT(rel_w2, rw2T, 64, 128, 64);
  cvtT(rel_w3, rw3T, 128, 256, 128);

  // Fused PointNets (dynamic LDS: weights + per-wave staging = 136,960 B)
  const size_t PN_SMEM =
      (size_t)(128 * 64 + 256 * 128 + 64 * 32 + 8 * 16 * 128 + 8 * 16 * 32) * 2 +
      (size_t)(64 + 128 + 256 + 8 * 256) * 4;
  pointnet_kernel<3, 512><<<dim3(N_NODES), dim3(256), PN_SMEM, stream>>>(
      obj_points, ow1T, obj_b1, ow2T, obj_b2, ow3T, obj_b3, xh);
  pointnet_kernel<4, 256><<<dim3(N_EDGES), dim3(256), PN_SMEM, stream>>>(
      rel_points, rw1T, rel_b1, rw2T, rel_b2, rw3T, rel_b3, eh);

  // degree counts (shared by both layers)
  zero_kernel<<<dim3(2), dim3(256), 0, stream>>>(cnt, N_NODES);
  deg_kernel<<<dim3(16), dim3(256), 0, stream>>>(ei, cnt);

  for (int l = 0; l < 2; ++l) {
    gather_kernel<<<dim3(N_EDGES), dim3(256), 0, stream>>>(xh, eh, ei, tbuf);
    gemm(tbuf, g_tw1T + (size_t)l * 512 * 768, g_tb1 + l * 512,
         nullptr, h1h, N_EDGES, HG, DD3, 1);
    gemm(h1h, g_tw2T + (size_t)l * 768 * 512, g_tb2 + l * 768,
         hout, nullptr, N_EDGES, DD3, HG, 0);
    zero_kernel<<<dim3(N_NODES), dim3(256), 0, stream>>>(pooled, N_NODES * DD);
    scatter_kernel<<<dim3(N_EDGES), dim3(256), 0, stream>>>(hout, ei, eh, pooled);
    avg_kernel<<<dim3(N_NODES), dim3(256), 0, stream>>>(pooled, cnt, avgh);
    gemm(avgh, g_nw1T + (size_t)l * 512 * 256, g_nb1 + l * 512,
         nullptr, nh, N_NODES, HG, DD, 1);
    gemm(nh, g_nw2T + (size_t)l * 256 * 512, g_nb2 + l * 256,
         nullptr, xh, N_NODES, DD, HG, 0);
  }

  // object head -> log_softmax
  gemm(xh, oc_w1T, oc_b1, nullptr, och1, N_NODES, 512, 256, 1);
  gemm(och1, oc_w2T, oc_b2, nullptr, och2, N_NODES, 256, 512, 1);
  gemm(och2, oc_w3T, oc_b3, oclog, nullptr, N_NODES, NOC, 256, 0);
  logsoftmax_kernel<<<dim3(N_NODES), dim3(256), 0, stream>>>(oclog, (float*)d_out);

  // relation head -> sigmoid
  gemm(eh, rc_w1T, rc_b1, nullptr, rch1, N_EDGES, 512, 256, 1);
  gemm(rch1, rc_w2T, rc_b2, nullptr, rch2, N_EDGES, 256, 512, 1);
  gemm(rch2, rc_w3T, rc_b3, rclog, nullptr, N_EDGES, NRC, 256, 0);
  sigmoid_kernel<<<dim3((N_EDGES * NRC + 255) / 256), dim3(256), 0, stream>>>(
      rclog, (float*)d_out + (size_t)N_NODES * NOC, N_EDGES * NRC);
}
